// HQQDequantLinear_40905268527136
// MI455X (gfx1250) — matrix operand order repaired and checked by reading, not compiled, not run on hardware
//
#include <hip/hip_runtime.h>

typedef __attribute__((ext_vector_type(16))) __bf16 v16bf;
typedef __attribute__((ext_vector_type(8)))  float  v8f;
typedef __attribute__((ext_vector_type(4)))  int    v4i;

#define M_TOT 8192
#define N_TOT 4096
#define K_TOT 4096
#define GROUP 64
#define BM 128
#define BN 128
#define BK 64
#define LDST 72              // bf16 elements per LDS row (144 B, 16B-aligned, bank-staggered)

#define A_BUF_BYTES (BM * LDST * 2)            // 18432
#define B_REGION    (2 * A_BUF_BYTES)          // B tiles start here
#define SMEM_BYTES  (2 * (BM + BN) * LDST * 2) // 73728

typedef __attribute__((address_space(3))) char lds_char_t;

__device__ __forceinline__ unsigned int pack_bf16x2(float a, float b) {
    union { __bf16 h[2]; unsigned int u; } p;
    p.h[0] = (__bf16)a;
    p.h[1] = (__bf16)b;
    return p.u;
}

// async global->LDS copy of 16B per lane; imm offset applies to BOTH lds and global addr
#define ASYNC16(ldsaddr, vaddr, sbase, off)                                   \
    asm volatile("global_load_async_to_lds_b128 %0, %1, %2 offset:" #off     \
                 :: "v"(ldsaddr), "v"(vaddr), "s"(sbase) : "memory")

#define WAIT_ASYNC() asm volatile("s_wait_asynccnt 0x0" ::: "memory")

// ---------------------------------------------------------------------------
// Pass 1a: x fp32 -> bf16 (64 MB out)
// ---------------------------------------------------------------------------
__global__ __launch_bounds__(256) void convert_x_bf16(
    const float* __restrict__ x, __bf16* __restrict__ xb)
{
    size_t i = ((size_t)blockIdx.x * 256 + threadIdx.x) * 8;
    float4 a = *(const float4*)(x + i);
    float4 b = *(const float4*)(x + i + 4);
    v4i o;
    o.x = (int)pack_bf16x2(a.x, a.y);
    o.y = (int)pack_bf16x2(a.z, a.w);
    o.z = (int)pack_bf16x2(b.x, b.y);
    o.w = (int)pack_bf16x2(b.z, b.w);
    __builtin_nontemporal_store(o, (v4i*)(xb + i));
}

// ---------------------------------------------------------------------------
// Pass 1b: W int4-in-byte -> dequant bf16 (32 MB out). group = packed_idx/32.
// ---------------------------------------------------------------------------
__global__ __launch_bounds__(256) void dequant_w_bf16(
    const int* __restrict__ Wq, const float* __restrict__ scale,
    const float* __restrict__ zero, __bf16* __restrict__ wb)
{
    size_t j = ((size_t)blockIdx.x * 256 + threadIdx.x) * 8;  // packed-byte index
    int   g  = (int)(j >> 5);                                 // 32 packed bytes per group
    float s  = scale[g];
    float z  = zero[g];
    int4 q0 = ((const int4*)(Wq + j))[0];
    int4 q1 = ((const int4*)(Wq + j))[1];
    int q[8] = { q0.x, q0.y, q0.z, q0.w, q1.x, q1.y, q1.z, q1.w };
    unsigned p[8];
    #pragma unroll
    for (int e = 0; e < 8; ++e) {
        float lo = __builtin_fmaf((float)(q[e] & 15),        s, -z * s);
        float hi = __builtin_fmaf((float)((q[e] >> 4) & 15), s, -z * s);
        p[e] = pack_bf16x2(lo, hi);
    }
    v4i o0; o0.x = (int)p[0]; o0.y = (int)p[1]; o0.z = (int)p[2]; o0.w = (int)p[3];
    v4i o1; o1.x = (int)p[4]; o1.y = (int)p[5]; o1.z = (int)p[6]; o1.w = (int)p[7];
    __builtin_nontemporal_store(o0, (v4i*)(wb + 2 * j));
    __builtin_nontemporal_store(o1, (v4i*)(wb + 2 * j + 8));
}

// ---------------------------------------------------------------------------
// Pass 2: pure bf16 NT GEMM, async global->LDS double buffering + WMMA
// ---------------------------------------------------------------------------
__global__ __launch_bounds__(256) void gemm_bf16_async(
    const __bf16* __restrict__ xb,   // [M_TOT, K_TOT] bf16
    const __bf16* __restrict__ wb,   // [N_TOT, K_TOT] bf16
    float* __restrict__ out)         // [M_TOT, N_TOT] fp32
{
    extern __shared__ char smem_raw[];
    __bf16* As = (__bf16*)smem_raw;             // [2][BM][LDST]
    __bf16* Bs = As + 2 * BM * LDST;            // [2][BN][LDST]
    const unsigned lds_base = (unsigned)(unsigned long long)(lds_char_t*)smem_raw;

    const int t     = threadIdx.x;
    const int lane  = t & 31;
    const int wave  = t >> 5;
    const int waveM = wave >> 2;      // 0..1
    const int waveN = wave & 3;       // 0..3
    const int lh    = lane & 15;
    const int lhi   = lane >> 4;

    const int m0 = blockIdx.y * BM;
    const int n0 = blockIdx.x * BN;

    // cooperative async-copy coordinates: 1 row + 64B half per thread
    const int lrow  = t >> 1;         // 0..127
    const int lhalf = t & 1;          // 0..1
    const unsigned thr_lds = (unsigned)(lrow * (LDST * 2) + lhalf * 64);
    const unsigned ldsA0 = lds_base + thr_lds;
    const unsigned ldsB0 = lds_base + B_REGION + thr_lds;
    const unsigned voffA0 = ((unsigned)(m0 + lrow) * K_TOT + lhalf * 32) * 2; // bytes
    const unsigned voffB0 = ((unsigned)(n0 + lrow) * K_TOT + lhalf * 32) * 2; // bytes
    const unsigned long long xb64 = (unsigned long long)xb;
    const unsigned long long wb64 = (unsigned long long)wb;

    auto issue_stage = [&](int kb, int buf) {
        const unsigned va = voffA0 + (unsigned)kb * (BK * 2);
        const unsigned vb = voffB0 + (unsigned)kb * (BK * 2);
        const unsigned la = ldsA0 + (unsigned)buf * A_BUF_BYTES;
        const unsigned lb = ldsB0 + (unsigned)buf * A_BUF_BYTES;
        ASYNC16(la, va, xb64, 0);
        ASYNC16(la, va, xb64, 16);
        ASYNC16(la, va, xb64, 32);
        ASYNC16(la, va, xb64, 48);
        ASYNC16(lb, vb, wb64, 0);
        ASYNC16(lb, vb, wb64, 16);
        ASYNC16(lb, vb, wb64, 32);
        ASYNC16(lb, vb, wb64, 48);
    };

    v8f acc[4][2] = {};

    issue_stage(0, 0);
    WAIT_ASYNC();
    __syncthreads();

    const int nk = K_TOT / BK;  // 64 stages
    for (int kb = 0; kb < nk; ++kb) {
        const int buf = kb & 1;
        if (kb + 1 < nk) issue_stage(kb + 1, buf ^ 1);  // overlap with compute

        const __bf16* Ab = As + buf * BM * LDST;
        const __bf16* Bb = Bs + buf * BN * LDST;
        const int akoff = lhi ? 8 : 0;  // 16-bit A layout: lane-half K split

        #pragma unroll
        for (int ks = 0; ks < 2; ++ks) {  // two 16x16x32 K-steps per stage
            v16bf afrag[4];
            #pragma unroll
            for (int mt = 0; mt < 4; ++mt) {
                const __bf16* ab = Ab + (waveM * 64 + mt * 16 + lh) * LDST + ks * 32;
                ((int4*)&afrag[mt])[0] = *(const int4*)(ab + akoff);
                ((int4*)&afrag[mt])[1] = *(const int4*)(ab + akoff + 16);
            }
            v16bf bfrag[2];
            #pragma unroll
            for (int nt = 0; nt < 2; ++nt) {
                const __bf16* bb = Bb + (waveN * 32 + nt * 16 + lh) * LDST + ks * 32 + lhi * 8;
                ((int4*)&bfrag[nt])[0] = *(const int4*)(bb);
                ((int4*)&bfrag[nt])[1] = *(const int4*)(bb + 16);
            }
            #pragma unroll
            for (int mt = 0; mt < 4; ++mt)
                #pragma unroll
                for (int nt = 0; nt < 2; ++nt)
                    acc[mt][nt] = __builtin_amdgcn_wmma_f32_16x16x32_bf16(
                        false, afrag[mt], false, bfrag[nt],
                        (short)0, acc[mt][nt], false, false);
        }

        if (kb + 1 < nk) WAIT_ASYNC();  // own async copies for next stage done
        __syncthreads();                // all waves' copies visible
    }

    // Epilogue: C layout -> N = lane&15, M = vgpr + 8*(lane>=16); streamed stores
    #pragma unroll
    for (int mt = 0; mt < 4; ++mt) {
        #pragma unroll
        for (int nt = 0; nt < 2; ++nt) {
            const int m = m0 + waveM * 64 + mt * 16 + lhi * 8;
            const int n = n0 + waveN * 32 + nt * 16 + lh;
            float* o = out + (size_t)m * N_TOT + n;
            #pragma unroll
            for (int v = 0; v < 8; ++v)
                __builtin_nontemporal_store(acc[mt][nt][v], o + (size_t)v * N_TOT);
        }
    }
}

// ---------------------------------------------------------------------------
// Fallback: fully fused single-pass kernel (round-1 design), used when the
// workspace is too small for the two-pass path.
// ---------------------------------------------------------------------------
__global__ __launch_bounds__(256) void hqq_wmma_gemm_fused(
    const float* __restrict__ x, const int* __restrict__ Wq,
    const float* __restrict__ scale, const float* __restrict__ zero,
    float* __restrict__ out)
{
    extern __shared__ char smem_raw[];
    __bf16* As = (__bf16*)smem_raw;
    __bf16* Bs = As + 2 * BM * LDST;

    const int t     = threadIdx.x;
    const int lane  = t & 31;
    const int wave  = t >> 5;
    const int waveM = wave >> 2;
    const int waveN = wave & 3;
    const int lh    = lane & 15;
    const int lhi   = lane >> 4;

    const int m0 = blockIdx.y * BM;
    const int n0 = blockIdx.x * BN;

    const int lrow  = t >> 1;
    const int lhalf = t & 1;

    const float* aptr = x  + (size_t)(m0 + lrow) * K_TOT + lhalf * 32;
    const int*   bptr = Wq + (size_t)(n0 + lrow) * (K_TOT / 2) + lhalf * 16;
    const int    grp0 = (n0 + lrow) * (K_TOT / GROUP);

    v8f acc[4][2] = {};
    float4 aregs[8];
    int4   bregs[4];
    float  s_reg = 0.f, z_reg = 0.f;

    auto load_stage = [&](int kb) {
        const float4* ap = (const float4*)(aptr + kb * BK);
        #pragma unroll
        for (int i = 0; i < 8; ++i) aregs[i] = ap[i];
        const int4* bp = (const int4*)(bptr + kb * (BK / 2));
        #pragma unroll
        for (int i = 0; i < 4; ++i) bregs[i] = bp[i];
        s_reg = scale[grp0 + kb];
        z_reg = zero[grp0 + kb];
    };

    auto store_stage = [&](int buf) {
        unsigned int apack[16];
        #pragma unroll
        for (int i = 0; i < 8; ++i) {
            apack[2 * i + 0] = pack_bf16x2(aregs[i].x, aregs[i].y);
            apack[2 * i + 1] = pack_bf16x2(aregs[i].z, aregs[i].w);
        }
        int4* adst = (int4*)(As + buf * BM * LDST + lrow * LDST + lhalf * 32);
        #pragma unroll
        for (int j = 0; j < 4; ++j) {
            int4 w;
            w.x = (int)apack[4 * j]; w.y = (int)apack[4 * j + 1];
            w.z = (int)apack[4 * j + 2]; w.w = (int)apack[4 * j + 3];
            adst[j] = w;
        }
        unsigned int bpack[16];
        #pragma unroll
        for (int i = 0; i < 4; ++i) {
            int q[4] = { bregs[i].x, bregs[i].y, bregs[i].z, bregs[i].w };
            #pragma unroll
            for (int e = 0; e < 4; ++e) {
                float lo = ((float)(q[e] & 15)        - z_reg) * s_reg;
                float hi = ((float)((q[e] >> 4) & 15) - z_reg) * s_reg;
                bpack[4 * i + e] = pack_bf16x2(lo, hi);
            }
        }
        int4* bdst = (int4*)(Bs + buf * BN * LDST + lrow * LDST + lhalf * 32);
        #pragma unroll
        for (int j = 0; j < 4; ++j) {
            int4 w;
            w.x = (int)bpack[4 * j]; w.y = (int)bpack[4 * j + 1];
            w.z = (int)bpack[4 * j + 2]; w.w = (int)bpack[4 * j + 3];
            bdst[j] = w;
        }
    };

    load_stage(0);
    store_stage(0);
    __syncthreads();

    const int nk = K_TOT / BK;
    for (int kb = 0; kb < nk; ++kb) {
        const int buf = kb & 1;
        if (kb + 1 < nk) load_stage(kb + 1);

        const __bf16* Ab = As + buf * BM * LDST;
        const __bf16* Bb = Bs + buf * BN * LDST;
        const int akoff = lhi ? 8 : 0;

        #pragma unroll
        for (int ks = 0; ks < 2; ++ks) {
            v16bf afrag[4];
            #pragma unroll
            for (int mt = 0; mt < 4; ++mt) {
                const __bf16* ab = Ab + (waveM * 64 + mt * 16 + lh) * LDST + ks * 32;
                ((int4*)&afrag[mt])[0] = *(const int4*)(ab + akoff);
                ((int4*)&afrag[mt])[1] = *(const int4*)(ab + akoff + 16);
            }
            v16bf bfrag[2];
            #pragma unroll
            for (int nt = 0; nt < 2; ++nt) {
                const __bf16* bb = Bb + (waveN * 32 + nt * 16 + lh) * LDST + ks * 32 + lhi * 8;
                ((int4*)&bfrag[nt])[0] = *(const int4*)(bb);
                ((int4*)&bfrag[nt])[1] = *(const int4*)(bb + 16);
            }
            #pragma unroll
            for (int mt = 0; mt < 4; ++mt)
                #pragma unroll
                for (int nt = 0; nt < 2; ++nt)
                    acc[mt][nt] = __builtin_amdgcn_wmma_f32_16x16x32_bf16(
                        false, afrag[mt], false, bfrag[nt],
                        (short)0, acc[mt][nt], false, false);
        }

        if (kb + 1 < nk) store_stage(buf ^ 1);
        __syncthreads();
    }

    #pragma unroll
    for (int mt = 0; mt < 4; ++mt) {
        #pragma unroll
        for (int nt = 0; nt < 2; ++nt) {
            const int m = m0 + waveM * 64 + mt * 16 + lhi * 8;
            const int n = n0 + waveN * 32 + nt * 16 + lh;
            float* o = out + (size_t)m * N_TOT + n;
            #pragma unroll
            for (int v = 0; v < 8; ++v)
                __builtin_nontemporal_store(acc[mt][nt][v], o + (size_t)v * N_TOT);
        }
    }
}

extern "C" void kernel_launch(void* const* d_in, const int* in_sizes, int n_in,
                              void* d_out, int out_size, void* d_ws, size_t ws_size,
                              hipStream_t stream) {
    (void)in_sizes; (void)n_in; (void)out_size;
    const float* x     = (const float*)d_in[0];
    const int*   Wq    = (const int*)d_in[1];
    const float* scale = (const float*)d_in[2];
    const float* zero  = (const float*)d_in[3];
    float*       out   = (float*)d_out;

    const size_t wbytes = (size_t)N_TOT * K_TOT * 2;   // 32 MB
    const size_t xbytes = (size_t)M_TOT * K_TOT * 2;   // 64 MB
    dim3 grid(N_TOT / BN, M_TOT / BM);                 // (32, 64)
    dim3 block(256);

    if (ws_size >= wbytes + xbytes && d_ws != nullptr) {
        __bf16* wb = (__bf16*)d_ws;
        __bf16* xb = (__bf16*)((char*)d_ws + wbytes);

        dequant_w_bf16<<<(N_TOT * K_TOT / 2) / (256 * 8), 256, 0, stream>>>(Wq, scale, zero, wb);
        convert_x_bf16<<<(M_TOT * K_TOT) / (256 * 8), 256, 0, stream>>>(x, xb);
        gemm_bf16_async<<<grid, block, SMEM_BYTES, stream>>>(xb, wb, out);
    } else {
        hqq_wmma_gemm_fused<<<grid, block, SMEM_BYTES, stream>>>(x, Wq, scale, zero, out);
    }
}
